// FastSelfAttention_85847806312764
// MI455X (gfx1250) — matrix operand order repaired and checked by reading, not compiled, not run on hardware
//
#include <hip/hip_runtime.h>
#include <math.h>

typedef _Float16 v16h __attribute__((ext_vector_type(16)));
typedef _Float16 v8h  __attribute__((ext_vector_type(8)));
typedef float    v8f  __attribute__((ext_vector_type(8)));

#define H_    16
#define B_    4
#define L_    4096
#define D_    1024
#define DK    64
#define M_    256
#define BH    (B_*H_)
#define NROWS (B_*L_)    // 16384

// ---------------- WMMA fragment helpers (ISA 7.12.2 layouts, wave32) --------

__device__ __forceinline__ v16h load_a16(const _Float16* __restrict__ a, int lda, int lane) {
  int m = lane & 15, half = lane >> 4;
  const _Float16* p = a + (size_t)m * lda + half * 8;
  union { v16h v; v8h h[2]; } f;
  f.h[0] = *(const v8h*)(p);
  f.h[1] = *(const v8h*)(p + 16);
  return f.v;
}

// B: 32x16 f16 supplied as Bt row-major [N x K] with leading dim ldb.
__device__ __forceinline__ v16h load_b16(const _Float16* __restrict__ bt, int ldb, int lane) {
  int n = lane & 15, half = lane >> 4;
  const _Float16* p = bt + (size_t)n * ldb + half * 8;
  union { v16h v; v8h h[2]; } f;
  f.h[0] = *(const v8h*)(p);
  f.h[1] = *(const v8h*)(p + 16);
  return f.v;
}

__device__ __forceinline__ v8f wmma_f16(v16h a, v16h b, v8f c) {
  return __builtin_amdgcn_wmma_f32_16x16x32_f16(false, a, false, b, (short)0, c, false, false);
}

// ---------------- CDNA5 async global->LDS copy (ASYNCcnt path) --------------

__device__ __forceinline__ void async_copy_b128(unsigned lds_byte_addr,
                                                const _Float16* __restrict__ gsrc) {
  asm volatile("global_load_async_to_lds_b128 %0, %1, off"
               :: "v"(lds_byte_addr), "v"(gsrc) : "memory");
}

__device__ __forceinline__ void wait_asynccnt0() {
  asm volatile("s_wait_asynccnt 0x0" ::: "memory");
}

// ---------------- block-cooperative pipelined GEMM core (MT=2) --------------
// 256 threads = 8 waves. Each wave computes 32 rows x 64 cols (2x4 C tiles).
// All 8 waves share one B panel [64 cols x K] (Bt row-major, ldb).
// B K-slices (64x32 f16 = 4KB) staged in LDS via async copies, double-buffered.
// K loop manually unrolled by 2 (dedicated buffer/A registers per phase, so the
// back-edge keeps accumulators in fixed registers -> no register-rotation movs).
// Requires (K/32) even, which holds for all call sites (8, 32, 128).

__device__ __forceinline__ void gemm_block2(const _Float16* __restrict__ abase, int lda,
                                            const _Float16* __restrict__ btbase, int ldb,
                                            int K, _Float16* ldsB, v8f acc[2][4]) {
  const int tid  = threadIdx.x;
  const int lane = tid & 31;
  const int m = lane & 15, half = lane >> 4;

  // async staging map: thread t copies 16B; col n = t>>2, segment s = t&3
  const int cn = tid >> 2, cs = tid & 3;
  const _Float16* gsrc = btbase + (size_t)cn * ldb + cs * 8;
  const unsigned dst0 = (unsigned)(uintptr_t)ldsB + (unsigned)(cn * 32 + cs * 8) * 2u;
  const unsigned dst1 = dst0 + 64u * 32u * 2u;

  const _Float16* arow0 = abase + (size_t)m * lda + half * 8;
  const _Float16* arow1 = arow0 + (size_t)16 * lda;

  union F { v16h v; v8h h[2]; };
  F a0e, a1e, a0o, a1o;

  // prologue: stage k-slice 0 into buffer 0, prefetch first A fragments
  async_copy_b128(dst0, gsrc);
  a0e.h[0] = *(const v8h*)(arow0);  a0e.h[1] = *(const v8h*)(arow0 + 16);
  a1e.h[0] = *(const v8h*)(arow1);  a1e.h[1] = *(const v8h*)(arow1 + 16);

  const int nsteps = K >> 5;
  for (int i = 0; i < nsteps; i += 2) {
    // ---- even phase: consume buf0, stage slice i+1 into buf1 ----
    wait_asynccnt0();
    __syncthreads();
    async_copy_b128(dst1, gsrc + (size_t)(i + 1) * 32);
    {
      const _Float16* p0 = arow0 + (size_t)(i + 1) * 32;
      const _Float16* p1 = arow1 + (size_t)(i + 1) * 32;
      a0o.h[0] = *(const v8h*)(p0); a0o.h[1] = *(const v8h*)(p0 + 16);
      a1o.h[0] = *(const v8h*)(p1); a1o.h[1] = *(const v8h*)(p1 + 16);
    }
#pragma unroll
    for (int j = 0; j < 4; ++j) {
      v16h b = load_b16(ldsB + (size_t)(16 * j) * 32, 32, lane);
      acc[0][j] = wmma_f16(a0e.v, b, acc[0][j]);
      acc[1][j] = wmma_f16(a1e.v, b, acc[1][j]);
    }
    // ---- odd phase: consume buf1, stage slice i+2 into buf0 ----
    wait_asynccnt0();
    __syncthreads();
    if (i + 2 < nsteps) {
      async_copy_b128(dst0, gsrc + (size_t)(i + 2) * 32);
      const _Float16* p0 = arow0 + (size_t)(i + 2) * 32;
      const _Float16* p1 = arow1 + (size_t)(i + 2) * 32;
      a0e.h[0] = *(const v8h*)(p0); a0e.h[1] = *(const v8h*)(p0 + 16);
      a1e.h[0] = *(const v8h*)(p1); a1e.h[1] = *(const v8h*)(p1 + 16);
    }
#pragma unroll
    for (int j = 0; j < 4; ++j) {
      v16h b = load_b16(ldsB + 64 * 32 + (size_t)(16 * j) * 32, 32, lane);
      acc[0][j] = wmma_f16(a0o.v, b, acc[0][j]);
      acc[1][j] = wmma_f16(a1o.v, b, acc[1][j]);
    }
  }
}

// ---------------- conversion -----------------------------------------------

__global__ void cvt4_f32_to_f16(const float* __restrict__ in, _Float16* __restrict__ out, int n4) {
  int i = blockIdx.x * blockDim.x + threadIdx.x;
  if (i >= n4) return;
  float4 v = ((const float4*)in)[i];
  union { _Float16 e[4]; unsigned long long u; } p;
  p.e[0] = (_Float16)v.x; p.e[1] = (_Float16)v.y;
  p.e[2] = (_Float16)v.z; p.e[3] = (_Float16)v.w;
  *(unsigned long long*)(out + (size_t)i * 4) = p.u;
}

// ---------------- QKV projection: out = x @ W^T + b -------------------------
// grid (64, 16): block tile = 256 rows x 64 cols.
// mode 0: row-major f16 out. mode 1 (V): transposed per head -> vT[b,h,d,l]

__global__ void __launch_bounds__(256)
gemm_qkv(const _Float16* __restrict__ xh, const _Float16* __restrict__ Wh,
         const float* __restrict__ bias,
         _Float16* __restrict__ outN, _Float16* __restrict__ outVT, int mode) {
  __shared__ _Float16 ldsB[2 * 64 * 32];
  int lane = threadIdx.x & 31;
  int wave = threadIdx.x >> 5;
  int row0 = blockIdx.x * 256 + wave * 32;
  int col0 = blockIdx.y * 64;
  v8f acc[2][4] = {};
  gemm_block2(xh + (size_t)row0 * D_, D_, Wh + (size_t)col0 * D_, D_, D_, ldsB, acc);

  int nl = lane & 15, half = lane >> 4;
#pragma unroll
  for (int rt = 0; rt < 2; ++rt) {
#pragma unroll
    for (int j = 0; j < 4; ++j) {
      int c = col0 + 16 * j + nl;
      float bv = bias[c];
#pragma unroll
      for (int v = 0; v < 8; ++v) {
        int r = row0 + rt * 16 + v + 8 * half;
        float val = acc[rt][j][v] + bv;
        if (mode == 0) {
          outN[(size_t)r * D_ + c] = (_Float16)val;
        } else {
          int b = r >> 12, l = r & (L_ - 1);
          int h = c >> 6,  d = c & 63;
          outVT[(((size_t)(b * H_ + h) * DK + d) * L_) + l] = (_Float16)val;
        }
      }
    }
  }
}

// ---------------- per-(row,head) squared-norm: sumsq/(2*sqrt(dk)) -----------

__global__ void rownorm(const _Float16* __restrict__ qh, float* __restrict__ nrm, int total) {
  int i = blockIdx.x * blockDim.x + threadIdx.x;
  if (i >= total) return;
  int row = i >> 4, h = i & 15;
  const _Float16* p = qh + (size_t)row * D_ + h * DK;
  float s = 0.f;
#pragma unroll
  for (int c = 0; c < DK; c += 8) {
    v8h v = *(const v8h*)(p + c);
#pragma unroll
    for (int e = 0; e < 8; ++e) { float f = (float)v[e]; s += f * f; }
  }
  nrm[i] = s * 0.0625f;
}

// ---------------- phi = (exp(x.orf^T/dk^0.25 - norm) + eps)/sqrt(m) ---------
// B (orf) is 32KB and L2/WGP$-resident: direct loads, K = 64 (2 wmma steps).

__global__ void phi_kernel(const _Float16* __restrict__ qh, const _Float16* __restrict__ orfh,
                           const float* __restrict__ nrm,
                           _Float16* __restrict__ phiN, _Float16* __restrict__ phiT, int mode) {
  int lane = threadIdx.x & 31;
  int wid  = blockIdx.x * 8 + (threadIdx.x >> 5);
  int bh = blockIdx.y;
  int b = bh >> 4, h = bh & 15;
  int l0 = (wid >> 3) * 16;
  int m0 = (wid & 7) * 32;
  const _Float16* abase = qh + (size_t)(b * L_ + l0) * D_ + h * DK;
  v8f acc[2] = {};
#pragma unroll
  for (int k0 = 0; k0 < DK; k0 += 32) {
    v16h a = load_a16(abase + k0, D_, lane);
#pragma unroll
    for (int t = 0; t < 2; ++t) {
      v16h bb = load_b16(orfh + (size_t)(m0 + 16 * t) * DK + k0, DK, lane);
      acc[t] = wmma_f16(a, bb, acc[t]);
    }
  }
  int nl = lane & 15, half = lane >> 4;
  const float kscale = 0.35355339059327373f;   // 64^(-1/4)
#pragma unroll
  for (int t = 0; t < 2; ++t) {
    int m = m0 + 16 * t + nl;
#pragma unroll
    for (int v = 0; v < 8; ++v) {
      int l = l0 + v + 8 * half;
      float nr = nrm[(size_t)(b * L_ + l) * H_ + h];
      float val = (expf(acc[t][v] * kscale - nr) + 1e-6f) * 0.0625f;
      if (mode == 0) phiN[((size_t)bh * L_ + l) * M_ + m] = (_Float16)val;
      else           phiT[((size_t)bh * M_ + m) * L_ + l] = (_Float16)val;
    }
  }
}

// ---------------- kv = k_phi^T @ v : [256 x 64] per head, K = L = 4096 ------
// grid (1, 64): one block per head; each wave owns 32 m-rows, shares vT panel.
// output stored transposed: kvT[bh, d, m]

__global__ void __launch_bounds__(256)
kv_gemm(const _Float16* __restrict__ kphiT, const _Float16* __restrict__ vT,
        _Float16* __restrict__ kvT) {
  __shared__ _Float16 ldsB[2 * 64 * 32];
  int lane = threadIdx.x & 31;
  int wave = threadIdx.x >> 5;
  int bh = blockIdx.y;
  int m0 = wave * 32;
  v8f acc[2][4] = {};
  gemm_block2(kphiT + ((size_t)bh * M_ + m0) * L_, L_,
              vT + (size_t)bh * DK * L_, L_, L_, ldsB, acc);

  int nl = lane & 15, half = lane >> 4;
#pragma unroll
  for (int rt = 0; rt < 2; ++rt) {
#pragma unroll
    for (int j = 0; j < 4; ++j) {
      int d = 16 * j + nl;
      union { v8h v; _Float16 e[8]; } pk;
#pragma unroll
      for (int v = 0; v < 8; ++v) pk.e[v] = (_Float16)acc[rt][j][v];   // m contiguous
      *(v8h*)(kvT + ((size_t)bh * DK + d) * M_ + m0 + rt * 16 + 8 * half) = pk.v;
    }
  }
}

// ---------------- ksum[bh,m] = sum_l k_phi ---------------------------------

__global__ void ksum_kernel(const _Float16* __restrict__ kphiT, float* __restrict__ ksum) {
  int lane = threadIdx.x & 31;
  int row = blockIdx.x * 8 + (threadIdx.x >> 5);
  const _Float16* p = kphiT + (size_t)row * L_;
  float s = 0.f;
  for (int it = 0; it < L_ / 256; ++it) {
    v8h v = *(const v8h*)(p + it * 256 + lane * 8);
#pragma unroll
    for (int e = 0; e < 8; ++e) s += (float)v[e];
  }
#pragma unroll
  for (int off = 16; off; off >>= 1) s += __shfl_xor(s, off, 32);
  if (lane == 0) ksum[row] = s;
}

// ---------------- den[bh,l] = q_phi[l,:] . ksum ----------------------------

__global__ void den_kernel(const _Float16* __restrict__ qphi, const float* __restrict__ ksum,
                           float* __restrict__ den) {
  int lane = threadIdx.x & 31;
  int row = blockIdx.x * 8 + (threadIdx.x >> 5);
  int bh = row >> 12;
  v8h qv = *(const v8h*)(qphi + (size_t)row * M_ + lane * 8);
  const float* kp = ksum + (size_t)bh * M_ + lane * 8;
  float s = 0.f;
#pragma unroll
  for (int e = 0; e < 8; ++e) s += (float)qv[e] * kp[e];
#pragma unroll
  for (int off = 16; off; off >>= 1) s += __shfl_xor(s, off, 32);
  if (lane == 0) den[row] = s;
}

// ---------------- num = q_phi @ kv, fused /den, concat heads ----------------
// grid (16, 64): each wave owns 32 l-rows; waves share kvT[bh] panel. K = 256.

__global__ void __launch_bounds__(256)
num_gemm(const _Float16* __restrict__ qphi, const _Float16* __restrict__ kvT,
         const float* __restrict__ den, _Float16* __restrict__ attn) {
  __shared__ _Float16 ldsB[2 * 64 * 32];
  int lane = threadIdx.x & 31;
  int lt = blockIdx.x * 8 + (threadIdx.x >> 5);
  int bh = blockIdx.y;
  int b = bh >> 4, h = bh & 15;
  int l0 = lt * 32;
  v8f acc[2][4] = {};
  gemm_block2(qphi + ((size_t)bh * L_ + l0) * M_, M_,
              kvT + (size_t)bh * DK * M_, M_, M_, ldsB, acc);

  int nl = lane & 15, half = lane >> 4;
#pragma unroll
  for (int rt = 0; rt < 2; ++rt) {
#pragma unroll
    for (int j = 0; j < 4; ++j) {
      int d = 16 * j + nl;
#pragma unroll
      for (int v = 0; v < 8; ++v) {
        int l = l0 + rt * 16 + v + 8 * half;
        float val = acc[rt][j][v] / den[(size_t)bh * L_ + l];
        attn[(size_t)(b * L_ + l) * D_ + h * DK + d] = (_Float16)val;
      }
    }
  }
}

// ---------------- final projection: out = attn @ Wo^T + bo (f32) ------------

__global__ void __launch_bounds__(256)
gemm_out(const _Float16* __restrict__ attn, const _Float16* __restrict__ Woh,
         const float* __restrict__ bo, float* __restrict__ out) {
  __shared__ _Float16 ldsB[2 * 64 * 32];
  int lane = threadIdx.x & 31;
  int wave = threadIdx.x >> 5;
  int row0 = blockIdx.x * 256 + wave * 32;
  int col0 = blockIdx.y * 64;
  v8f acc[2][4] = {};
  gemm_block2(attn + (size_t)row0 * D_, D_, Woh + (size_t)col0 * D_, D_, D_, ldsB, acc);

  int nl = lane & 15, half = lane >> 4;
#pragma unroll
  for (int rt = 0; rt < 2; ++rt) {
#pragma unroll
    for (int j = 0; j < 4; ++j) {
      int c = col0 + 16 * j + nl;
      float bv = bo[c];
#pragma unroll
      for (int v = 0; v < 8; ++v)
        out[(size_t)(row0 + rt * 16 + v + 8 * half) * D_ + c] = acc[rt][j][v] + bv;
    }
  }
}

// ---------------------------------------------------------------------------

extern "C" void kernel_launch(void* const* d_in, const int* in_sizes, int n_in,
                              void* d_out, int out_size, void* d_ws, size_t ws_size,
                              hipStream_t stream) {
  (void)in_sizes; (void)n_in; (void)out_size; (void)ws_size;
  const float* x   = (const float*)d_in[0];
  const float* Wq  = (const float*)d_in[1];
  const float* bq  = (const float*)d_in[2];
  const float* Wk  = (const float*)d_in[3];
  const float* bk  = (const float*)d_in[4];
  const float* Wv  = (const float*)d_in[5];
  const float* bv  = (const float*)d_in[6];
  const float* Wo  = (const float*)d_in[7];
  const float* bo  = (const float*)d_in[8];
  const float* orf = (const float*)d_in[9];
  float* out = (float*)d_out;

  char* ws = (char*)d_ws;
  size_t off = 0;
  auto alloc = [&](size_t bytes) -> void* {
    void* p = ws + off;
    off = (off + bytes + 255) & ~(size_t)255;
    return p;
  };

  _Float16* x_h    = (_Float16*)alloc((size_t)NROWS * D_ * 2);
  _Float16* wq_h   = (_Float16*)alloc((size_t)D_ * D_ * 2);
  _Float16* wk_h   = (_Float16*)alloc((size_t)D_ * D_ * 2);
  _Float16* wv_h   = (_Float16*)alloc((size_t)D_ * D_ * 2);
  _Float16* wo_h   = (_Float16*)alloc((size_t)D_ * D_ * 2);
  _Float16* orf_h  = (_Float16*)alloc((size_t)M_ * DK * 2);
  _Float16* q_h    = (_Float16*)alloc((size_t)NROWS * D_ * 2);
  _Float16* k_h    = (_Float16*)alloc((size_t)NROWS * D_ * 2);
  _Float16* v_hT   = (_Float16*)alloc((size_t)BH * DK * L_ * 2);
  float*    qn     = (float*)alloc((size_t)NROWS * H_ * 4);
  float*    kn     = (float*)alloc((size_t)NROWS * H_ * 4);
  _Float16* q_phi  = (_Float16*)alloc((size_t)BH * L_ * M_ * 2);
  _Float16* k_phiT = (_Float16*)alloc((size_t)BH * M_ * L_ * 2);
  _Float16* kvT    = (_Float16*)alloc((size_t)BH * DK * M_ * 2);
  float*    ksum   = (float*)alloc((size_t)BH * M_ * 4);
  float*    den    = (float*)alloc((size_t)BH * L_ * 4);
  _Float16* attn_h = (_Float16*)alloc((size_t)NROWS * D_ * 2);

  {
    int n4 = NROWS * D_ / 4;
    cvt4_f32_to_f16<<<(n4 + 255) / 256, 256, 0, stream>>>(x, x_h, n4);
    int w4 = D_ * D_ / 4;
    cvt4_f32_to_f16<<<(w4 + 255) / 256, 256, 0, stream>>>(Wq, wq_h, w4);
    cvt4_f32_to_f16<<<(w4 + 255) / 256, 256, 0, stream>>>(Wk, wk_h, w4);
    cvt4_f32_to_f16<<<(w4 + 255) / 256, 256, 0, stream>>>(Wv, wv_h, w4);
    cvt4_f32_to_f16<<<(w4 + 255) / 256, 256, 0, stream>>>(Wo, wo_h, w4);
    int o4 = M_ * DK / 4;
    cvt4_f32_to_f16<<<(o4 + 255) / 256, 256, 0, stream>>>(orf, orf_h, o4);
  }

  dim3 gproj(64, 16);
  gemm_qkv<<<gproj, 256, 0, stream>>>(x_h, wq_h, bq, q_h, (_Float16*)nullptr, 0);
  gemm_qkv<<<gproj, 256, 0, stream>>>(x_h, wk_h, bk, k_h, (_Float16*)nullptr, 0);
  gemm_qkv<<<gproj, 256, 0, stream>>>(x_h, wv_h, bv, (_Float16*)nullptr, v_hT, 1);

  rownorm<<<(NROWS * H_ + 255) / 256, 256, 0, stream>>>(q_h, qn, NROWS * H_);
  rownorm<<<(NROWS * H_ + 255) / 256, 256, 0, stream>>>(k_h, kn, NROWS * H_);

  phi_kernel<<<dim3(256, 64), 256, 0, stream>>>(q_h, orf_h, qn, q_phi, (_Float16*)nullptr, 0);
  phi_kernel<<<dim3(256, 64), 256, 0, stream>>>(k_h, orf_h, kn, (_Float16*)nullptr, k_phiT, 1);

  kv_gemm<<<dim3(1, 64), 256, 0, stream>>>(k_phiT, v_hT, kvT);
  ksum_kernel<<<BH * M_ / 8, 256, 0, stream>>>(k_phiT, ksum);
  den_kernel<<<BH * L_ / 8, 256, 0, stream>>>(q_phi, ksum, den);

  num_gemm<<<dim3(16, 64), 256, 0, stream>>>(q_phi, kvT, den, attn_h);

  gemm_out<<<dim3(64, 16), 256, 0, stream>>>(attn_h, wo_h, bo, out);
}
